// LSTMCells_37374805410181
// MI455X (gfx1250) — matrix operand order repaired and checked by reading, not compiled, not run on hardware
//
#include <hip/hip_runtime.h>
#include <hip/hip_bf16.h>

// ---------------------------------------------------------------------------
// LSTM-like recurrence, persistent multi-workgroup kernel for gfx1250 (MI455X)
//   B=64, T=512, F=256, H=512
//   step:  gates = [h|x_t] @ Wg^T + bg           (Wg = cat(W1..W4) : [2048,768])
//          c1 = sig(g0)*c + sig(g1)*tanh(g3)
//          h1 = tanh(c1 @ W5^T + b5) * sig(g2)
// 32 persistent workgroups; WG j owns hidden units [16j,16j+16).
// Weights live in LDS as bf16; matrix math = v_wmma_f32_16x16x32_bf16.
// ---------------------------------------------------------------------------

#define B_   64
#define T_   512
#define F_   256
#define H_   512
#define KDIM 768           // F + H
#define NWG  32
#define UPW  16            // hidden units per workgroup
#define THREADS 128        // 4 waves (wave32)

// Dynamic LDS layout (bytes) -- all offsets 16B aligned
#define OFF_WG    0                         // 64 x 768 bf16  = 98304
#define OFF_W5    98304                     // 16 x 512 bf16  = 16384
#define OFF_INP   114688                    // 64 x 768 bf16  = 98304 (phase B reuses as c-full 64x512)
#define OFF_GATES 212992                    // 4 x 64 x 16 f32 = 16384
#define OFF_C     229376                    // 64 x 16 f32     = 4096
#define OFF_BIAS  233472                    // 80 f32          = 320
#define LDS_TOTAL 233792

typedef __attribute__((ext_vector_type(16))) __bf16 v16bf;
typedef __attribute__((ext_vector_type(8)))  float  v8f;
typedef __attribute__((ext_vector_type(4)))  unsigned v4u;

union Frag { v16bf v; v4u q[2]; };

__device__ inline unsigned short f2bf(float f) {
  unsigned u = __float_as_uint(f);
  unsigned r = u + 0x7FFFu + ((u >> 16) & 1u);   // round-to-nearest-even
  return (unsigned short)(r >> 16);
}
__device__ inline float sigmf(float x)  { return 1.f / (1.f + __expf(-x)); }
__device__ inline float tanh_f(float x) { return 2.f / (1.f + __expf(-2.f * x)) - 1.f; }

// A matrix 16x32 bf16 (MxK): lane l holds row (l&15); element e -> K = 8*(l>=16) + e + 8*(e>=8).
// => two contiguous 16B chunks at kb and kb+16 elements, kb = 32*kt + 8*(l>=16).
__device__ inline v16bf load_a16(const unsigned short* base, int row, int stride, int kb) {
  Frag u;
  const unsigned short* p = base + row * stride + kb;
  u.q[0] = *(const v4u*)(p);
  u.q[1] = *(const v4u*)(p + 16);
  return u.v;
}
// B matrix 32x16 bf16 (KxN): lane l holds column (l&15); element e -> K = 8*(l>=16) + e + 8*(e>=8).
// => two contiguous 16B chunks at k0 and k0+16, k0 = 32*kt + 8*(l>=16) in a row-major [N][K] weight slab.
__device__ inline v16bf load_b16(const unsigned short* base, int row, int stride, int k0) {
  Frag u;
  const unsigned short* p = base + row * stride + k0;
  u.q[0] = *(const v4u*)(p);
  u.q[1] = *(const v4u*)(p + 16);
  return u.v;
}

__device__ inline v8f wmma_bf16(v16bf a, v16bf b, v8f c) {
  return __builtin_amdgcn_wmma_f32_16x16x32_bf16(false, a, false, b, (short)0, c, false, false);
}

__device__ inline void grid_barrier(unsigned* bar) {
  __threadfence();                 // make this thread's global stores agent-visible
  __syncthreads();
  if (threadIdx.x == 0) {
    unsigned gen  = __hip_atomic_load(&bar[1], __ATOMIC_ACQUIRE, __HIP_MEMORY_SCOPE_AGENT);
    unsigned prev = __hip_atomic_fetch_add(&bar[0], 1u, __ATOMIC_ACQ_REL, __HIP_MEMORY_SCOPE_AGENT);
    if (prev == NWG - 1) {
      __hip_atomic_store(&bar[0], 0u, __ATOMIC_RELAXED, __HIP_MEMORY_SCOPE_AGENT);
      __hip_atomic_fetch_add(&bar[1], 1u, __ATOMIC_RELEASE, __HIP_MEMORY_SCOPE_AGENT);
    } else {
      while (__hip_atomic_load(&bar[1], __ATOMIC_ACQUIRE, __HIP_MEMORY_SCOPE_AGENT) == gen) {
        __builtin_amdgcn_s_sleep(2);
      }
    }
  }
  __syncthreads();
}

__global__ void init_state_kernel(unsigned* p, int n_uints) {
  for (int i = blockIdx.x * blockDim.x + threadIdx.x; i < n_uints; i += gridDim.x * blockDim.x)
    p[i] = 0u;
}

__global__ __launch_bounds__(THREADS, 1)
void lstm_persistent(const float* __restrict__ x,
                     const float* __restrict__ W1, const float* __restrict__ b1,
                     const float* __restrict__ W2, const float* __restrict__ b2,
                     const float* __restrict__ W3, const float* __restrict__ b3,
                     const float* __restrict__ W4, const float* __restrict__ b4,
                     const float* __restrict__ W5, const float* __restrict__ b5,
                     float* __restrict__ out,
                     unsigned short* __restrict__ h_state,   // [64][512] bf16
                     unsigned short* __restrict__ c_shared,  // [64][512] bf16
                     unsigned* __restrict__ bar) {
  extern __shared__ char smem[];
  unsigned short* sWg   = (unsigned short*)(smem + OFF_WG);
  unsigned short* sW5   = (unsigned short*)(smem + OFF_W5);
  unsigned short* sInp  = (unsigned short*)(smem + OFF_INP);
  float*          sGates= (float*)(smem + OFF_GATES);
  float*          sC    = (float*)(smem + OFF_C);
  float*          sBias = (float*)(smem + OFF_BIAS);

  const int tid  = threadIdx.x;
  const int wv   = tid >> 5;          // wave 0..3
  const int lane = tid & 31;
  const int lr   = lane & 15;
  const int hs   = lane >> 4;         // half-select
  const int u0   = blockIdx.x * UPW;  // first hidden unit owned by this WG

  const float* Wg[4] = {W1, W2, W3, W4};
  const float* bg[4] = {b1, b2, b3, b4};

  // ---- one-time: stage bf16 weight slices into LDS ----
  for (int idx = tid; idx < 64 * KDIM; idx += THREADS) {
    int gr = idx / KDIM, k = idx - gr * KDIM;
    int g = gr >> 4, r = gr & 15;
    sWg[gr * KDIM + k] = f2bf(Wg[g][(size_t)(u0 + r) * KDIM + k]);
  }
  for (int idx = tid; idx < UPW * H_; idx += THREADS) {
    int r = idx / H_, k = idx - r * H_;
    sW5[r * H_ + k] = f2bf(W5[(size_t)(u0 + r) * H_ + k]);
  }
  if (tid < 64) sBias[tid] = bg[tid >> 4][u0 + (tid & 15)];
  if (tid < UPW) sBias[64 + tid] = b5[u0 + tid];
  for (int idx = tid; idx < B_ * UPW; idx += THREADS) sC[idx] = 0.f;
  grid_barrier(bar);   // also waits for init_state zeroing visibility across WGs

  for (int t = 0; t < T_; ++t) {
    // ================= phase A: build inp = [h | x_t] (bf16) =================
    {
      const unsigned* hs32 = (const unsigned*)h_state;        // [64][256] dwords
      for (int idx = tid; idx < B_ * (H_ / 2); idx += THREADS) {
        int b = idx >> 8, cp = idx & 255;
        *(unsigned*)&sInp[b * KDIM + 2 * cp] = hs32[b * 256 + cp];
      }
      for (int idx = tid; idx < B_ * (F_ / 2); idx += THREADS) {
        int b = idx >> 7, cp = idx & 127;
        const float* xp = x + ((size_t)b * T_ + t) * F_ + 2 * cp;
        unsigned lo = f2bf(xp[0]);
        unsigned hi = f2bf(xp[1]);
        *(unsigned*)&sInp[b * KDIM + H_ + 2 * cp] = lo | (hi << 16);
        if (t + 1 < T_) __builtin_prefetch(xp + F_, 0, 1);    // next timestep row
      }
    }
    __syncthreads();

    // ---- gate GEMM: wave wv computes gate wv's [64 x 16] block, K = 768 ----
    {
      v8f a0 = {0,0,0,0,0,0,0,0}, a1 = a0, a2 = a0, a3 = a0;
      for (int kt = 0; kt < KDIM / 32; ++kt) {
        v16bf bF = load_b16(sWg, wv * 16 + lr, KDIM, kt * 32 + 8 * hs);
        int kb = kt * 32 + 8 * hs;
        a0 = wmma_bf16(load_a16(sInp,  0 + lr, KDIM, kb), bF, a0);
        a1 = wmma_bf16(load_a16(sInp, 16 + lr, KDIM, kb), bF, a1);
        a2 = wmma_bf16(load_a16(sInp, 32 + lr, KDIM, kb), bF, a2);
        a3 = wmma_bf16(load_a16(sInp, 48 + lr, KDIM, kb), bF, a3);
      }
      float bias = sBias[wv * 16 + lr];
      #pragma unroll
      for (int v = 0; v < 8; ++v) {
        int m = v + hs * 8;
        sGates[(wv * 64 + ( 0 + m)) * 16 + lr] = a0[v] + bias;
        sGates[(wv * 64 + (16 + m)) * 16 + lr] = a1[v] + bias;
        sGates[(wv * 64 + (32 + m)) * 16 + lr] = a2[v] + bias;
        sGates[(wv * 64 + (48 + m)) * 16 + lr] = a3[v] + bias;
      }
    }
    __syncthreads();

    // ---- elementwise cell update on this WG's 16-unit slice ----
    for (int e = tid; e < B_ * UPW; e += THREADS) {
      int b = e >> 4, u = e & 15;
      float s1 = sigmf (sGates[(0 * 64 + b) * 16 + u]);
      float s2 = sigmf (sGates[(1 * 64 + b) * 16 + u]);
      float gg = tanh_f(sGates[(3 * 64 + b) * 16 + u]);
      float c1 = s1 * sC[e] + s2 * gg;
      sC[e] = c1;
      c_shared[b * H_ + u0 + u] = f2bf(c1);
    }
    grid_barrier(bar);   // publish c slices; all WGs now see full c1

    // ================= phase B: h1 = tanh(c1 @ W5^T + b5) * s3 =================
    {  // pull full c1 (bf16, [64][512]) into LDS, reusing the inp buffer
      const unsigned* cs32 = (const unsigned*)c_shared;
      unsigned* dst = (unsigned*)sInp;
      for (int idx = tid; idx < B_ * H_ / 2; idx += THREADS) dst[idx] = cs32[idx];
    }
    __syncthreads();

    {
      v8f acc = {0,0,0,0,0,0,0,0};
      for (int kt = 0; kt < H_ / 32; ++kt) {
        v16bf aF = load_a16(sInp, wv * 16 + lr, H_, kt * 32 + 8 * hs);
        v16bf bF = load_b16(sW5, lr, H_, kt * 32 + 8 * hs);
        acc = wmma_bf16(aF, bF, acc);
      }
      float bias = sBias[64 + lr];
      #pragma unroll
      for (int v = 0; v < 8; ++v) {
        int m = wv * 16 + v + hs * 8;                 // batch index
        float s3 = sigmf(sGates[(2 * 64 + m) * 16 + lr]);
        float hv = tanh_f(acc[v] + bias) * s3;
        out[((size_t)m * T_ + t) * H_ + u0 + lr] = hv;
        h_state[m * H_ + u0 + lr] = f2bf(hv);
      }
    }
    grid_barrier(bar);   // publish h before any WG starts step t+1
  }
}

extern "C" void kernel_launch(void* const* d_in, const int* in_sizes, int n_in,
                              void* d_out, int out_size, void* d_ws, size_t ws_size,
                              hipStream_t stream) {
  const float* x  = (const float*)d_in[0];
  const float* W1 = (const float*)d_in[1];  const float* b1 = (const float*)d_in[2];
  const float* W2 = (const float*)d_in[3];  const float* b2 = (const float*)d_in[4];
  const float* W3 = (const float*)d_in[5];  const float* b3 = (const float*)d_in[6];
  const float* W4 = (const float*)d_in[7];  const float* b4 = (const float*)d_in[8];
  const float* W5 = (const float*)d_in[9];  const float* b5 = (const float*)d_in[10];
  float* out = (float*)d_out;

  char* ws = (char*)d_ws;
  unsigned short* h_state  = (unsigned short*)(ws);            // 64*512*2 = 65536 B
  unsigned short* c_shared = (unsigned short*)(ws + 65536);    // 65536 B
  unsigned*       bar      = (unsigned*)(ws + 131072);         // 8 B

  // zero h0, c0 and the barrier (deterministic per call)
  int n_uints = (65536 + 65536 + 8) / 4;
  hipLaunchKernelGGL(init_state_kernel, dim3(32), dim3(256), 0, stream,
                     (unsigned*)d_ws, n_uints);

  hipFuncSetAttribute((const void*)lstm_persistent,
                      hipFuncAttributeMaxDynamicSharedMemorySize, LDS_TOTAL);
  hipLaunchKernelGGL(lstm_persistent, dim3(NWG), dim3(THREADS), LDS_TOTAL, stream,
                     x, W1, b1, W2, b2, W3, b3, W4, b4, W5, b5,
                     out, h_state, c_shared, bar);
}
